// SWS_Attention_76991583748440
// MI455X (gfx1250) — matrix operand order repaired and checked by reading, not compiled, not run on hardware
//
#include <hip/hip_runtime.h>
#include <hip/hip_bf16.h>

typedef __attribute__((ext_vector_type(16))) _Float16 v16h;
typedef __attribute__((ext_vector_type(8)))  _Float16 v8h;
typedef __attribute__((ext_vector_type(8)))  float    v8f;
typedef __attribute__((ext_vector_type(4)))  float    v4f;

#define BDIM 4
#define NTOK 2048
#define CDIM 768
#define HDS  12
#define HD   64

__device__ __forceinline__ v8f wmma16(v16h a, v16h b, v8f c) {
  return __builtin_amdgcn_wmma_f32_16x16x32_f16(false, a, false, b, (short)0, c, false, false);
}

// Async DMA: 16B per lane, global -> LDS, tracked by ASYNCcnt.
__device__ __forceinline__ void async_b128(unsigned int lds_off, unsigned long long gaddr) {
  asm volatile("global_load_async_to_lds_b128 %0, %1, off"
               :: "v"(lds_off), "v"(gaddr) : "memory");
}
__device__ __forceinline__ void wait_async0() {
  asm volatile("s_wait_asynccnt 0x0" ::: "memory");
}

// A fragment (16x32, 16-bit): lane l -> row m=l&15; K run = hi*16..+15 (contiguous)
template <typename PT>
__device__ __forceinline__ v16h load_a_frag(PT src, int ld, int lane) {
  int m = lane & 15, hi = lane >> 4;
  PT row = src + (size_t)m * ld;
  return *(const v16h*)(row + hi * 16);
}

// B fragment (32x16, 16-bit): lane l -> col n=l&15; K run = hi*16..+15 (contiguous).
// src is n-major with k contiguous (src[n][k] == Bmat[k][n]), stride ld.
template <typename PT>
__device__ __forceinline__ v16h load_b_frag(PT src, int ld, int lane) {
  int n = lane & 15, hi = lane >> 4;
  return *(const v16h*)(src + (size_t)n * ld + hi * 16);
}

// ---------------------------------------------------------------------------
// Kernel 1: QKV projection.  M=8192, K=768, Nout=2304.
// Epilogue: Q (scaled) and K head-major [B][H][N][64]; V TRANSPOSED [B][H][64][N].
// `which` (q/k/v slab) is uniform per block: 64 | 768, so no divergence.
// ---------------------------------------------------------------------------
__global__ __launch_bounds__(128) void sws_qkv_gemm(
    const float* __restrict__ x, const float* __restrict__ qkv_w,
    _Float16* __restrict__ qh, _Float16* __restrict__ kh, _Float16* __restrict__ vtw) {
  __shared__ _Float16 xs[64][32];
  __shared__ _Float16 wsm[64][32];
  const int t = threadIdx.x;
  const int lane = t & 31, w = t >> 5;
  const int nlo = lane & 15, hi = lane >> 4;
  const int Mblk = blockIdx.y * 64, Nblk = blockIdx.x * 64;
  const int mw = (w >> 1) * 32, nw = (w & 1) * 32;
  v8f c[2][2] = {};

  const int srow = t >> 1, skk = (t & 1) * 16;
  for (int kb = 0; kb < CDIM; kb += 32) {
    __syncthreads();
    {
      const float* xsrc = x + (size_t)(Mblk + srow) * CDIM + kb + skk;
      const float* wsrc = qkv_w + (size_t)(Nblk + srow) * CDIM + kb + skk;
#pragma unroll
      for (int i = 0; i < 16; i += 4) {
        v4f fx = *(const v4f*)(xsrc + i);
        v4f fw = *(const v4f*)(wsrc + i);
#pragma unroll
        for (int j = 0; j < 4; ++j) {
          xs[srow][skk + i + j]  = (_Float16)fx[j];
          wsm[srow][skk + i + j] = (_Float16)fw[j];
        }
      }
    }
    __syncthreads();
    v16h a0 = load_a_frag(&xs[mw][0], 32, lane);
    v16h a1 = load_a_frag(&xs[mw + 16][0], 32, lane);
    v16h b0 = load_b_frag(&wsm[nw][0], 32, lane);
    v16h b1 = load_b_frag(&wsm[nw + 16][0], 32, lane);
    c[0][0] = wmma16(a0, b0, c[0][0]);
    c[0][1] = wmma16(a0, b1, c[0][1]);
    c[1][0] = wmma16(a1, b0, c[1][0]);
    c[1][1] = wmma16(a1, b1, c[1][1]);
  }

  // ---- uniform epilogue: which slab (0=Q,1=K,2=V) is constant per block ----
  const int which = Nblk / CDIM;          // scalar (block-uniform)
  const int nrel  = Nblk - which * CDIM;  // column base inside the 768 slab

  if (which == 0) {
#pragma unroll
    for (int mi = 0; mi < 2; ++mi)
#pragma unroll
      for (int ni = 0; ni < 2; ++ni)
#pragma unroll
        for (int r = 0; r < 8; ++r) {
          int grow = Mblk + mw + mi * 16 + hi * 8 + r;
          int jj = nrel + nw + ni * 16 + nlo;
          int bh = (grow >> 11) * HDS + (jj >> 6);
          int ntok = grow & 2047, d = jj & 63;
          qh[(((size_t)bh * NTOK + ntok) << 6) + d] = (_Float16)(c[mi][ni][r] * 0.125f);
        }
  } else if (which == 1) {
#pragma unroll
    for (int mi = 0; mi < 2; ++mi)
#pragma unroll
      for (int ni = 0; ni < 2; ++ni)
#pragma unroll
        for (int r = 0; r < 8; ++r) {
          int grow = Mblk + mw + mi * 16 + hi * 8 + r;
          int jj = nrel + nw + ni * 16 + nlo;
          int bh = (grow >> 11) * HDS + (jj >> 6);
          int ntok = grow & 2047, d = jj & 63;
          kh[(((size_t)bh * NTOK + ntok) << 6) + d] = (_Float16)c[mi][ni][r];
        }
  } else {
#pragma unroll
    for (int mi = 0; mi < 2; ++mi)
#pragma unroll
      for (int ni = 0; ni < 2; ++ni)
#pragma unroll
        for (int r = 0; r < 8; ++r) {
          int grow = Mblk + mw + mi * 16 + hi * 8 + r;
          int jj = nrel + nw + ni * 16 + nlo;
          int bh = (grow >> 11) * HDS + (jj >> 6);
          int ntok = grow & 2047, d = jj & 63;
          vtw[((size_t)bh * HD + d) * NTOK + ntok] = (_Float16)c[mi][ni][r];  // transposed
        }
  }
}

// ---------------------------------------------------------------------------
// Kernel 2: flash attention.  grid = (N/64, B*H), 128 threads (4 waves).
// K tile [32][64] and V^T tile [64][32] are double-buffered in LDS via
// global_load_async_to_lds_b128; online softmax in fp32.
// ---------------------------------------------------------------------------
__global__ __launch_bounds__(128) void sws_flash_attn(
    const _Float16* __restrict__ qh, const _Float16* __restrict__ kh,
    const _Float16* __restrict__ vtw, _Float16* __restrict__ aout) {
  __shared__ _Float16 ks[2][32][64];   // K tile, row-major [key][d]
  __shared__ _Float16 vts[2][64][32];  // V^T tile, [d][key]
  __shared__ _Float16 ps[4][16][32];   // per-wave P re-layout buffer
  const int t = threadIdx.x, lane = t & 31, w = t >> 5;
  const int nlo = lane & 15, hi = lane >> 4;
  const int qt = blockIdx.x, bh = blockIdx.y;
  const size_t head = (size_t)bh * NTOK * HD;   // same elem count for all layouts
  const int qbase = qt * 64 + w * 16;

  v16h aq0 = load_a_frag(qh + head + (size_t)qbase * HD, HD, lane);
  v16h aq1 = load_a_frag(qh + head + (size_t)qbase * HD + 32, HD, lane);

  float mrun[8], lrun[8];
#pragma unroll
  for (int r = 0; r < 8; ++r) { mrun[r] = -1e30f; lrun[r] = 0.0f; }
  v8f o[4] = {};

  // Each thread DMAs 2 chunks of K tile + 2 chunks of V^T tile (16B each).
  auto stage = [&](int ib, int kt) {
#pragma unroll
    for (int u = 0; u < 2; ++u) {
      int c = t * 2 + u;                    // 0..255
      int kr = c >> 3, ko = (c & 7) * 8;    // K: 32 rows x 8 chunks
      async_b128((unsigned int)(size_t)&ks[ib][kr][ko],
                 (unsigned long long)(size_t)(kh + head + (size_t)(kt + kr) * HD + ko));
      int dr = c >> 2, doff = (c & 3) * 8;  // V^T: 64 rows x 4 chunks
      async_b128((unsigned int)(size_t)&vts[ib][dr][doff],
                 (unsigned long long)(size_t)(vtw + head + (size_t)dr * NTOK + kt + doff));
    }
  };

  stage(0, 0);
  for (int kt = 0; kt < NTOK; kt += 32) {
    const int ib = (kt >> 5) & 1;
    wait_async0();        // current tile landed in LDS
    __syncthreads();      // ...for every wave; also: prev tile fully consumed
    if (kt + 32 < NTOK) stage(ib ^ 1, kt + 32);   // prefetch next tile

    // S = Q * K^T  (16 q-rows x 32 keys), fp32 accum
    v8f s0 = {}, s1 = {};
#pragma unroll
    for (int cc = 0; cc < 2; ++cc) {
      v16h aqc = cc ? aq1 : aq0;
      v16h b0 = load_b_frag(&ks[ib][0][cc * 32], HD, lane);
      v16h b1 = load_b_frag(&ks[ib][16][cc * 32], HD, lane);
      s0 = wmma16(aqc, b0, s0);
      s1 = wmma16(aqc, b1, s1);
    }

    // online softmax; C-layout row m = hi*8 + r lives on 16 lanes of one half
#pragma unroll
    for (int r = 0; r < 8; ++r) {
      float t0 = s0[r], t1 = s1[r];
      float mx = fmaxf(t0, t1);
#pragma unroll
      for (int off = 8; off >= 1; off >>= 1) mx = fmaxf(mx, __shfl_xor(mx, off, 16));
      float mnew = fmaxf(mrun[r], mx);
      float corr = __expf(mrun[r] - mnew);
      float p0 = __expf(t0 - mnew);
      float p1 = __expf(t1 - mnew);
      float rs = p0 + p1;
#pragma unroll
      for (int off = 8; off >= 1; off >>= 1) rs += __shfl_xor(rs, off, 16);
      lrun[r] = lrun[r] * corr + rs;
      mrun[r] = mnew;
#pragma unroll
      for (int nb = 0; nb < 4; ++nb) o[nb][r] *= corr;
      ps[w][hi * 8 + r][nlo] = (_Float16)p0;
      ps[w][hi * 8 + r][16 + nlo] = (_Float16)p1;
    }
    asm volatile("s_wait_dscnt 0x0" ::: "memory");  // wave-local P reshuffle visible

    // O += P * V   (A = P from LDS, B = V^T tile from LDS)
    v16h ap = load_a_frag(&ps[w][0][0], 32, lane);
#pragma unroll
    for (int nb = 0; nb < 4; ++nb) {
      v16h bv = load_b_frag(&vts[ib][nb * 16][0], 32, lane);
      o[nb] = wmma16(ap, bv, o[nb]);
    }
  }

  // epilogue: O/l, write f16 into [B][N][H*64] layout for the proj GEMM
  const int b = bh / HDS, h = bh - b * HDS;
#pragma unroll
  for (int r = 0; r < 8; ++r) {
    float inv = 1.0f / lrun[r];
    int row = qbase + hi * 8 + r;
    _Float16* dst = aout + ((size_t)(b * NTOK + row)) * CDIM + h * HD + nlo;
#pragma unroll
    for (int nb = 0; nb < 4; ++nb)
      dst[nb * 16] = (_Float16)(o[nb][r] * inv);
  }
}

// ---------------------------------------------------------------------------
// Kernel 3: output projection + bias.  M=8192, K=768, Nout=768, fp32 out.
// X tile (already f16) is async-DMA'd to LDS; weight tile is cvt-staged.
// ---------------------------------------------------------------------------
__global__ __launch_bounds__(128) void sws_proj_gemm(
    const _Float16* __restrict__ ah, const float* __restrict__ proj_w,
    const float* __restrict__ proj_b, float* __restrict__ out) {
  __shared__ _Float16 xs[64][32];
  __shared__ _Float16 wsm[64][32];
  const int t = threadIdx.x;
  const int lane = t & 31, w = t >> 5;
  const int nlo = lane & 15, hi = lane >> 4;
  const int Mblk = blockIdx.y * 64, Nblk = blockIdx.x * 64;
  const int mw = (w >> 1) * 32, nw = (w & 1) * 32;
  v8f c[2][2] = {};

  const int srow = t >> 1, skk = (t & 1) * 16;
  for (int kb = 0; kb < CDIM; kb += 32) {
    __syncthreads();
    {
      const _Float16* xsrc = ah + (size_t)(Mblk + srow) * CDIM + kb + skk;
      async_b128((unsigned int)(size_t)&xs[srow][skk],
                 (unsigned long long)(size_t)xsrc);
      async_b128((unsigned int)(size_t)&xs[srow][skk + 8],
                 (unsigned long long)(size_t)(xsrc + 8));
      const float* wsrc = proj_w + (size_t)(Nblk + srow) * CDIM + kb + skk;
#pragma unroll
      for (int i = 0; i < 16; i += 4) {
        v4f fw = *(const v4f*)(wsrc + i);
#pragma unroll
        for (int j = 0; j < 4; ++j) wsm[srow][skk + i + j] = (_Float16)fw[j];
      }
    }
    wait_async0();
    __syncthreads();
    v16h a0 = load_a_frag(&xs[mw][0], 32, lane);
    v16h a1 = load_a_frag(&xs[mw + 16][0], 32, lane);
    v16h b0 = load_b_frag(&wsm[nw][0], 32, lane);
    v16h b1 = load_b_frag(&wsm[nw + 16][0], 32, lane);
    c[0][0] = wmma16(a0, b0, c[0][0]);
    c[0][1] = wmma16(a0, b1, c[0][1]);
    c[1][0] = wmma16(a1, b0, c[1][0]);
    c[1][1] = wmma16(a1, b1, c[1][1]);
  }

#pragma unroll
  for (int mi = 0; mi < 2; ++mi)
#pragma unroll
    for (int ni = 0; ni < 2; ++ni) {
      int j = Nblk + nw + ni * 16 + nlo;
      float bias = proj_b[j];
#pragma unroll
      for (int r = 0; r < 8; ++r) {
        int grow = Mblk + mw + mi * 16 + hi * 8 + r;
        out[(size_t)grow * CDIM + j] = c[mi][ni][r] + bias;
      }
    }
}

extern "C" void kernel_launch(void* const* d_in, const int* in_sizes, int n_in,
                              void* d_out, int out_size, void* d_ws, size_t ws_size,
                              hipStream_t stream) {
  const float* x      = (const float*)d_in[0];
  const float* qkv_w  = (const float*)d_in[1];
  const float* proj_w = (const float*)d_in[2];
  const float* proj_b = (const float*)d_in[3];
  float* out = (float*)d_out;

  // workspace: 4 f16 tensors of B*H*N*64 = 6,291,456 elems each (~50.3 MB total)
  const size_t perT = (size_t)BDIM * HDS * NTOK * HD;
  _Float16* qh  = (_Float16*)d_ws;
  _Float16* kh  = qh + perT;
  _Float16* vtw = kh + perT;   // V stored transposed per head: [B][H][64][N]
  _Float16* ah  = vtw + perT;

  dim3 gQKV(3 * CDIM / 64, (BDIM * NTOK) / 64);      // (36, 128)
  sws_qkv_gemm<<<gQKV, 128, 0, stream>>>(x, qkv_w, qh, kh, vtw);

  dim3 gAttn(NTOK / 64, BDIM * HDS);                 // (32, 48)
  sws_flash_attn<<<gAttn, 128, 0, stream>>>(qh, kh, vtw, ah);

  dim3 gProj(CDIM / 64, (BDIM * NTOK) / 64);         // (12, 128)
  sws_proj_gemm<<<gProj, 128, 0, stream>>>(ah, proj_w, proj_b, out);
}
